// SepNet_90744069030465
// MI455X (gfx1250) — matrix operand order repaired and checked by reading, not compiled, not run on hardware
//
#include <hip/hip_runtime.h>
#include <hip/hip_bf16.h>

#define N_NODES 30000
#define E_EDGES 400000
#define F_INF   512
#define NHID    128
#define HEADS   4
#define NCLS    10
#define NSETS   5
#define BN_EPS  1e-5f

typedef __attribute__((ext_vector_type(16))) __bf16 v16bf;
typedef __attribute__((ext_vector_type(8)))  float  v8f;

__device__ __forceinline__ unsigned short f2bf_bits(float f) {
  unsigned u = __float_as_uint(f);
  unsigned r = u + 0x7FFFu + ((u >> 16) & 1u);   // round-to-nearest-even
  return (unsigned short)(r >> 16);
}
__device__ __forceinline__ __bf16 f2bf(float f) {
  unsigned short s = f2bf_bits(f);
  return __builtin_bit_cast(__bf16, s);
}
__device__ __forceinline__ float eluf(float x) { return x > 0.f ? x : (expf(x) - 1.f); }

__device__ __forceinline__ void atomicMaxF(float* addr, float value) {
  int* ai = (int*)addr;
  int old = __float_as_int(*addr);
  while (__int_as_float(old) < value) {
    int assumed = old;
    old = atomicCAS(ai, assumed, __float_as_int(value));
    if (old == assumed) break;
  }
}

// ---------------- elementwise helpers ----------------
__global__ void k_fill(float* __restrict__ p, float v, int n) {
  int i = blockIdx.x * blockDim.x + threadIdx.x;
  if (i < n) p[i] = v;
}
__global__ void k_copy(const float* __restrict__ in, float* __restrict__ out, int n) {
  int i = blockIdx.x * blockDim.x + threadIdx.x;
  if (i < n) out[i] = in[i];
}

// ---------------- B repack: f32 [K x Ncols] -> bf16 fragment-order ----------------
// Packed layout: [tk][tn][lane][e], tk = K/32 tiles, tn = Ncols/16 tiles.
// Lane L of a wave needs B[tk*32 + 8*(L>>4) + (e&7) + 16*(e>>3)][tn*16 + (L&15)] at element e,
// so each lane's 16 bf16 (32 bytes) are contiguous -> 2x global_load_b128.
__global__ void k_pack_B(const float* __restrict__ W, unsigned short* __restrict__ out,
                         int K, int Ncols) {
  int p = blockIdx.x * blockDim.x + threadIdx.x;
  if (p >= K * Ncols) return;
  int nTn  = Ncols >> 4;
  int tile = p >> 9;          // 512 elems per (tk,tn) tile
  int rem  = p & 511;
  int lane = rem >> 4;
  int e    = rem & 15;
  int tn   = tile % nTn;
  int tk   = tile / nTn;
  int k = tk * 32 + ((lane >> 4) << 3) + (e & 7) + ((e >> 3) << 4);
  int n = tn * 16 + (lane & 15);
  out[p] = f2bf_bits(W[(size_t)k * Ncols + n]);
}

// ---------------- WMMA bf16 GEMM ----------------
// C[M x Ncols] = op(A[M x K]) @ B[K x Ncols] (+bias) (+BN, accumulate)
// 96 threads = 3 waves; each wave computes a 16x64 strip via 4 accumulators.
// B is pre-packed in fragment order (see k_pack_B).
// ACT: 0 = identity, 1 = elu(a + abias[k])   EPI: 0 = store acc+bias, 1 = BN then atomicAdd
template<int ACT, int EPI>
__global__ void k_wmma_gemm(const float* __restrict__ A, const unsigned short* __restrict__ Bp,
                            const float* __restrict__ bias, const float* __restrict__ abias,
                            float* __restrict__ C, int K, int Ncols,
                            const float* __restrict__ bnm, const float* __restrict__ bnv,
                            const float* __restrict__ bng, const float* __restrict__ bnb) {
  const int lane = threadIdx.x & 31;
  const int wave = threadIdx.x >> 5;
  const int row_base = (blockIdx.y * 3 + wave) * 16;
  const int col_base = blockIdx.x * 64;
  const int half = lane >> 4;
  const int m = row_base + (lane & 15);
  const int nTn = Ncols >> 4;
  const int tn0 = col_base >> 4;

  v8f acc[4];
#pragma unroll
  for (int t = 0; t < 4; ++t) acc[t] = (v8f){0.f,0.f,0.f,0.f,0.f,0.f,0.f,0.f};

  const float* arow = A + (size_t)m * K;

  for (int kb = 0; kb < K; kb += 32) {
    // ---- A fragment (16x32 bf16): K groups 8*half+e and 16+8*half+e, wide loads
    const int k0 = kb + half * 8;
    const float4* ap = (const float4*)(arow + k0);
    float4 q0 = ap[0], q1 = ap[1];           // K = k0..k0+7
    float4 q2 = ap[4], q3 = ap[5];           // K = k0+16..k0+23
    if (ACT) {
      const float4* bp4 = (const float4*)(abias + k0);
      float4 u0 = bp4[0], u1 = bp4[1], u2 = bp4[4], u3 = bp4[5];
      q0.x = eluf(q0.x + u0.x); q0.y = eluf(q0.y + u0.y); q0.z = eluf(q0.z + u0.z); q0.w = eluf(q0.w + u0.w);
      q1.x = eluf(q1.x + u1.x); q1.y = eluf(q1.y + u1.y); q1.z = eluf(q1.z + u1.z); q1.w = eluf(q1.w + u1.w);
      q2.x = eluf(q2.x + u2.x); q2.y = eluf(q2.y + u2.y); q2.z = eluf(q2.z + u2.z); q2.w = eluf(q2.w + u2.w);
      q3.x = eluf(q3.x + u3.x); q3.y = eluf(q3.y + u3.y); q3.z = eluf(q3.z + u3.z); q3.w = eluf(q3.w + u3.w);
    }
    v16bf a;
    a[0]=f2bf(q0.x); a[1]=f2bf(q0.y); a[2]=f2bf(q0.z); a[3]=f2bf(q0.w);
    a[4]=f2bf(q1.x); a[5]=f2bf(q1.y); a[6]=f2bf(q1.z); a[7]=f2bf(q1.w);
    a[8]=f2bf(q2.x); a[9]=f2bf(q2.y); a[10]=f2bf(q2.z); a[11]=f2bf(q2.w);
    a[12]=f2bf(q3.x); a[13]=f2bf(q3.y); a[14]=f2bf(q3.z); a[15]=f2bf(q3.w);

    // ---- B fragments: contiguous 32B per lane in packed layout
    const int tk = kb >> 5;
    const unsigned short* btile = Bp + (((size_t)tk * nTn + tn0) << 9) + (lane << 4);
    v16bf b0 = *(const v16bf*)(btile);
    v16bf b1 = *(const v16bf*)(btile + 512);
    v16bf b2 = *(const v16bf*)(btile + 1024);
    v16bf b3 = *(const v16bf*)(btile + 1536);

    acc[0] = __builtin_amdgcn_wmma_f32_16x16x32_bf16(false, a, false, b0, (short)0, acc[0], false, false);
    acc[1] = __builtin_amdgcn_wmma_f32_16x16x32_bf16(false, a, false, b1, (short)0, acc[1], false, false);
    acc[2] = __builtin_amdgcn_wmma_f32_16x16x32_bf16(false, a, false, b2, (short)0, acc[2], false, false);
    acc[3] = __builtin_amdgcn_wmma_f32_16x16x32_bf16(false, a, false, b3, (short)0, acc[3], false, false);
  }

  // ---- epilogue: D element r = C[row_base + r + 8*half][col]
  const int ncol = lane & 15;
#pragma unroll
  for (int t = 0; t < 4; ++t) {
    const int col = col_base + 16 * t + ncol;
    const float bcol = bias ? bias[col] : 0.f;
#pragma unroll
    for (int r = 0; r < 8; ++r) {
      const int row = row_base + r + 8 * half;
      float v = acc[t][r] + bcol;
      if (EPI == 1) {
        v = (v - bnm[col]) * rsqrtf(bnv[col] + BN_EPS) * bng[col] + bnb[col];
        atomicAdd(&C[(size_t)row * Ncols + col], v);
      } else {
        C[(size_t)row * Ncols + col] = v;
      }
    }
  }
}

// ---------------- sparse scatter: out[dst] += w * (abs?)in[src], 128 feats ----------------
__global__ void k_sp(const float* __restrict__ in, const int* __restrict__ src,
                     const int* __restrict__ dst, const float* __restrict__ w,
                     float* __restrict__ out, int nE, int doAbs) {
  int t = blockIdx.x * blockDim.x + threadIdx.x;
  int edge = t >> 5;
  if (edge >= nE) return;
  int lane = t & 31;
  int s = src[edge], d = dst[edge];
  float wv = w[edge];
  float4 v = ((const float4*)(in + (size_t)s * NHID))[lane];
  if (doAbs) { v.x = fabsf(v.x); v.y = fabsf(v.y); v.z = fabsf(v.z); v.w = fabsf(v.w); }
  float* op = out + (size_t)d * NHID + lane * 4;
  atomicAdd(op + 0, wv * v.x);
  atomicAdd(op + 1, wv * v.y);
  atomicAdd(op + 2, wv * v.z);
  atomicAdd(op + 3, wv * v.w);
}

// ---------------- GAT attention pieces ----------------
__global__ void k_att_proj(const float* __restrict__ hh, const float* __restrict__ att_src,
                           const float* __restrict__ att_dst, float* __restrict__ a_s,
                           float* __restrict__ a_d) {
  int idx = blockIdx.x * blockDim.x + threadIdx.x;   // n*H + h
  if (idx >= N_NODES * HEADS) return;
  int n = idx >> 2, h = idx & 3;
  const float* hp = hh + (size_t)n * (HEADS * NHID) + h * NHID;
  const float* as = att_src + h * NHID;
  const float* ad = att_dst + h * NHID;
  float s = 0.f, d = 0.f;
#pragma unroll 4
  for (int c = 0; c < NHID; ++c) { float v = hp[c]; s += v * as[c]; d += v * ad[c]; }
  a_s[idx] = s; a_d[idx] = d;
}

__global__ void k_edge_logits(const int* __restrict__ esrc, const int* __restrict__ edst,
                              const float* __restrict__ a_s, const float* __restrict__ a_d,
                              float* __restrict__ ebuf, float* __restrict__ mbuf) {
  int idx = blockIdx.x * blockDim.x + threadIdx.x;   // (E+N)*H
  if (idx >= (E_EDGES + N_NODES) * HEADS) return;
  int ed = idx >> 2, h = idx & 3;
  int s, d;
  if (ed < E_EDGES) { s = esrc[ed]; d = edst[ed]; } else { s = d = ed - E_EDGES; }
  float e = a_s[s * HEADS + h] + a_d[d * HEADS + h];
  e = e > 0.f ? e : 0.2f * e;                        // leaky_relu(0.2)
  ebuf[idx] = e;
  atomicMaxF(&mbuf[d * HEADS + h], e);
}

__global__ void k_edge_exp(const int* __restrict__ esrc, const int* __restrict__ edst,
                           const float* __restrict__ ebuf, const float* __restrict__ mbuf,
                           float* __restrict__ exbuf, float* __restrict__ den) {
  int idx = blockIdx.x * blockDim.x + threadIdx.x;
  if (idx >= (E_EDGES + N_NODES) * HEADS) return;
  int ed = idx >> 2, h = idx & 3;
  int d;
  if (ed < E_EDGES) { d = edst[ed]; } else { d = ed - E_EDGES; }
  float ex = expf(ebuf[idx] - mbuf[d * HEADS + h]);
  exbuf[idx] = ex;
  atomicAdd(&den[d * HEADS + h], ex);
}

__global__ void k_edge_aggr(const int* __restrict__ esrc, const int* __restrict__ edst,
                            const float* __restrict__ hh, const float* __restrict__ exbuf,
                            const float* __restrict__ den, float* __restrict__ gout) {
  long long t = (long long)blockIdx.x * blockDim.x + threadIdx.x;
  int ed = (int)(t >> 7);
  if (ed >= E_EDGES + N_NODES) return;
  int q = (int)(t & 127);                            // q*4 in [0,512)
  int s, d;
  if (ed < E_EDGES) { s = esrc[ed]; d = edst[ed]; } else { s = d = ed - E_EDGES; }
  int h = q >> 5;
  float alpha = exbuf[ed * HEADS + h] / (den[d * HEADS + h] + 1e-16f);
  float4 v = ((const float4*)(hh + (size_t)s * (HEADS * NHID)))[q];
  float* op = gout + (size_t)d * (HEADS * NHID) + q * 4;
  atomicAdd(op + 0, alpha * v.x);
  atomicAdd(op + 1, alpha * v.y);
  atomicAdd(op + 2, alpha * v.z);
  atomicAdd(op + 3, alpha * v.w);
}

// ---------------- output head ----------------
__global__ void k_bn0_elu(const float* __restrict__ x2, const float* __restrict__ g,
                          const float* __restrict__ b, const float* __restrict__ mm,
                          const float* __restrict__ vv, float* __restrict__ zact) {
  int idx = blockIdx.x * blockDim.x + threadIdx.x;
  if (idx >= N_NODES * NHID) return;
  int k = idx & (NHID - 1);
  float z = (x2[idx] - mm[k]) * rsqrtf(vv[k] + BN_EPS) * g[k] + b[k];
  zact[idx] = eluf(z);
}

__global__ void k_out_logits(const float* __restrict__ zact, const float* __restrict__ W,
                             const float* __restrict__ b, float* __restrict__ logits) {
  int idx = blockIdx.x * blockDim.x + threadIdx.x;   // n*NCLS + c
  if (idx >= N_NODES * NCLS) return;
  int n = idx / NCLS, c = idx - n * NCLS;
  const float* zp = zact + (size_t)n * NHID;
  float acc = b[c];
#pragma unroll 4
  for (int k = 0; k < NHID; ++k) acc += zp[k] * W[k * NCLS + c];
  logits[idx] = acc;
}

__global__ void k_logsoftmax(const float* __restrict__ logits, float* __restrict__ out) {
  int n = blockIdx.x * blockDim.x + threadIdx.x;
  if (n >= N_NODES) return;
  const float* lp = logits + (size_t)n * NCLS;
  float mx = lp[0];
#pragma unroll
  for (int c = 1; c < NCLS; ++c) mx = fmaxf(mx, lp[c]);
  float ss = 0.f;
#pragma unroll
  for (int c = 0; c < NCLS; ++c) ss += expf(lp[c] - mx);
  float lse = mx + logf(ss);
#pragma unroll
  for (int c = 0; c < NCLS; ++c) out[(size_t)n * NCLS + c] = lp[c] - lse;
}

// ---------------- host orchestration ----------------
static inline size_t alignup(size_t x) { return (x + 255) & ~(size_t)255; }

extern "C" void kernel_launch(void* const* d_in, const int* in_sizes, int n_in,
                              void* d_out, int out_size, void* d_ws, size_t ws_size,
                              hipStream_t stream) {
  (void)in_sizes; (void)n_in; (void)out_size; (void)ws_size;
  const float* x        = (const float*)d_in[0];
  const int*   eidx     = (const int*)d_in[1];
  const int*   sidx     = (const int*)d_in[2];
  const float* sw       = (const float*)d_in[3];
  const float* lin_W    = (const float*)d_in[4];
  const float* lin_b    = (const float*)d_in[5];
  const float* gat_W    = (const float*)d_in[6];
  const float* att_src  = (const float*)d_in[7];
  const float* att_dst  = (const float*)d_in[8];
  const float* gat_b    = (const float*)d_in[9];
  const float* mlp_W    = (const float*)d_in[10];
  const float* mlp_b    = (const float*)d_in[11];
  const float* bn_g     = (const float*)d_in[12];
  const float* bn_b     = (const float*)d_in[13];
  const float* bn_m     = (const float*)d_in[14];
  const float* bn_v     = (const float*)d_in[15];
  const float* bn0_g    = (const float*)d_in[16];
  const float* bn0_b    = (const float*)d_in[17];
  const float* bn0_m    = (const float*)d_in[18];
  const float* bn0_v    = (const float*)d_in[19];
  const float* out_W    = (const float*)d_in[20];
  const float* out_b    = (const float*)d_in[21];
  float* out = (float*)d_out;

  // carve workspace
  char* ws = (char*)d_ws;
  size_t off = 0;
  auto carve = [&](size_t bytes) -> void* { void* p = ws + off; off += alignup(bytes); return p; };
  const size_t NH = (size_t)N_NODES * NHID;           // 3.84M floats
  const size_t NHH = (size_t)N_NODES * HEADS * NHID;  // 15.36M floats
  const int ETOT = E_EDGES + N_NODES;

  float* h    = (float*)carve(NH * 4);
  float* tmp  = (float*)carve(NH * 4);
  float* xs[NSETS];
  for (int i = 0; i < NSETS; ++i) xs[i] = (float*)carve(NH * 4);
  float* hh   = (float*)carve(NHH * 4);
  float* gout = (float*)carve(NHH * 4);
  float* a_s  = (float*)carve((size_t)N_NODES * HEADS * 4);
  float* a_d  = (float*)carve((size_t)N_NODES * HEADS * 4);
  float* mbuf = (float*)carve((size_t)N_NODES * HEADS * 4);
  float* den  = (float*)carve((size_t)N_NODES * HEADS * 4);
  float* ebuf = (float*)carve((size_t)ETOT * HEADS * 4);
  float* exbuf= (float*)carve((size_t)ETOT * HEADS * 4);
  float* x2   = (float*)carve(NH * 4);
  float* zact = (float*)carve(NH * 4);
  float* logits = (float*)carve((size_t)N_NODES * NCLS * 4);
  unsigned short* WbfLin = (unsigned short*)carve((size_t)F_INF * NHID * 2);
  unsigned short* WbfGat = (unsigned short*)carve((size_t)NSETS * NHID * HEADS * NHID * 2);
  unsigned short* WbfMlp = (unsigned short*)carve((size_t)NSETS * HEADS * NHID * NHID * 2);

  const int* esrc = eidx;
  const int* edst = eidx + E_EDGES;

  auto g1 = [](size_t n, int bs) { return dim3((unsigned)((n + bs - 1) / bs)); };

  // repack weights f32 -> bf16 fragment-order
  k_pack_B<<<g1(F_INF * NHID, 256), 256, 0, stream>>>(lin_W, WbfLin, F_INF, NHID);
  for (int i = 0; i < NSETS; ++i) {
    k_pack_B<<<g1(NHID * HEADS * NHID, 256), 256, 0, stream>>>(
        gat_W + (size_t)i * NHID * HEADS * NHID, WbfGat + (size_t)i * NHID * HEADS * NHID,
        NHID, HEADS * NHID);
    k_pack_B<<<g1(HEADS * NHID * NHID, 256), 256, 0, stream>>>(
        mlp_W + (size_t)i * HEADS * NHID * NHID, WbfMlp + (size_t)i * HEADS * NHID * NHID,
        HEADS * NHID, NHID);
  }

  // h = x @ lin_W + lin_b   (M=30000, K=512, N=128)
  k_wmma_gemm<0, 0><<<dim3(NHID / 64, N_NODES / 48), 96, 0, stream>>>(
      x, WbfLin, lin_b, nullptr, h, F_INF, NHID, nullptr, nullptr, nullptr, nullptr);

  // sparse sets: xs[0] = sp0(h); xs[i>=1] = sp0(|sp_i(h)|)
  k_fill<<<g1(NH, 256), 256, 0, stream>>>(xs[0], 0.f, (int)NH);
  k_sp<<<g1((size_t)E_EDGES * 32, 256), 256, 0, stream>>>(
      h, sidx + 0 * 2 * E_EDGES, sidx + (0 * 2 + 1) * E_EDGES, sw + 0, xs[0], E_EDGES, 0);
  for (int i = 1; i < NSETS; ++i) {
    k_fill<<<g1(NH, 256), 256, 0, stream>>>(tmp, 0.f, (int)NH);
    k_sp<<<g1((size_t)E_EDGES * 32, 256), 256, 0, stream>>>(
        h, sidx + (i * 2 + 0) * E_EDGES, sidx + (i * 2 + 1) * E_EDGES, sw + (size_t)i * E_EDGES, tmp, E_EDGES, 0);
    k_fill<<<g1(NH, 256), 256, 0, stream>>>(xs[i], 0.f, (int)NH);
    k_sp<<<g1((size_t)E_EDGES * 32, 256), 256, 0, stream>>>(
        tmp, sidx + 0 * 2 * E_EDGES, sidx + (0 * 2 + 1) * E_EDGES, sw + 0, xs[i], E_EDGES, 1);
  }

  // stage2 accumulator starts at h
  k_copy<<<g1(NH, 256), 256, 0, stream>>>(h, x2, (int)NH);

  for (int i = 0; i < NSETS; ++i) {
    // hh = xs[i] @ gat_W[i]   (M=30000, K=128, N=512)
    k_wmma_gemm<0, 0><<<dim3(HEADS * NHID / 64, N_NODES / 48), 96, 0, stream>>>(
        xs[i], WbfGat + (size_t)i * NHID * HEADS * NHID, nullptr, nullptr, hh,
        NHID, HEADS * NHID, nullptr, nullptr, nullptr, nullptr);
    // attention logits
    k_att_proj<<<g1((size_t)N_NODES * HEADS, 256), 256, 0, stream>>>(
        hh, att_src + (size_t)i * HEADS * NHID, att_dst + (size_t)i * HEADS * NHID, a_s, a_d);
    k_fill<<<g1((size_t)N_NODES * HEADS, 256), 256, 0, stream>>>(mbuf, -3.0e38f, N_NODES * HEADS);
    k_fill<<<g1((size_t)N_NODES * HEADS, 256), 256, 0, stream>>>(den, 0.f, N_NODES * HEADS);
    k_fill<<<g1(NHH, 256), 256, 0, stream>>>(gout, 0.f, (int)NHH);
    k_edge_logits<<<g1((size_t)ETOT * HEADS, 256), 256, 0, stream>>>(esrc, edst, a_s, a_d, ebuf, mbuf);
    k_edge_exp<<<g1((size_t)ETOT * HEADS, 256), 256, 0, stream>>>(esrc, edst, ebuf, mbuf, exbuf, den);
    k_edge_aggr<<<g1((size_t)ETOT * 128, 256), 256, 0, stream>>>(esrc, edst, hh, exbuf, den, gout);
    // x2 += BN_i( elu(gout + gat_b[i]) @ mlp_W[i] + mlp_b[i] )   (M=30000, K=512, N=128)
    k_wmma_gemm<1, 1><<<dim3(NHID / 64, N_NODES / 48), 96, 0, stream>>>(
        gout, WbfMlp + (size_t)i * HEADS * NHID * NHID, mlp_b + (size_t)i * NHID,
        gat_b + (size_t)i * HEADS * NHID, x2, HEADS * NHID, NHID,
        bn_m + (size_t)i * NHID, bn_v + (size_t)i * NHID, bn_g + (size_t)i * NHID, bn_b + (size_t)i * NHID);
  }

  // head: bn0 -> elu -> @out_W + out_b -> log_softmax
  k_bn0_elu<<<g1(NH, 256), 256, 0, stream>>>(x2, bn0_g, bn0_b, bn0_m, bn0_v, zact);
  k_out_logits<<<g1((size_t)N_NODES * NCLS, 256), 256, 0, stream>>>(zact, out_W, out_b, logits);
  k_logsoftmax<<<g1(N_NODES, 256), 256, 0, stream>>>(logits, out);
}
